// ResidualVQ_54778012893241
// MI455X (gfx1250) — matrix operand order repaired and checked by reading, not compiled, not run on hardware
//
#include <hip/hip_runtime.h>

#define B_    16
#define T_    1500
#define DIM   512
#define NQ    8
#define KCB   1024
#define NTOK  (B_ * T_)      // 24000 tokens
#define NT    32             // tokens per workgroup (2 M-tiles of 16)
#define NWG   (NTOK / NT)    // 750 workgroups
#define LDSROW 520           // padded bf16 row stride (breaks bank conflicts)

typedef __attribute__((ext_vector_type(16))) __bf16        v16bf;
typedef __attribute__((ext_vector_type(8)))  float          v8f;
typedef __attribute__((ext_vector_type(8)))  unsigned short ushortx8;

union BFrag { v16bf v; ushortx8 h[2]; };

__device__ __forceinline__ unsigned short f32_to_bf16_rne(float f) {
    unsigned int u = __builtin_bit_cast(unsigned int, f);
    u += 0x7FFFu + ((u >> 16) & 1u);
    return (unsigned short)(u >> 16);
}

// ---------------------------------------------------------------- init
__global__ void vq_init(const float* __restrict__ x, const float* __restrict__ cb,
                        float* __restrict__ resid, unsigned short* __restrict__ cb16) {
    size_t stride = (size_t)gridDim.x * blockDim.x;
    size_t t0 = (size_t)blockIdx.x * blockDim.x + threadIdx.x;
    for (size_t i = t0; i < (size_t)NTOK * DIM; i += stride) resid[i] = x[i];
    for (size_t i = t0; i < (size_t)NQ * KCB * DIM; i += stride) cb16[i] = f32_to_bf16_rne(cb[i]);
}

// ---------------------------------------------------------------- c2 = ||codebook row||^2
__global__ void vq_c2(const float* __restrict__ cb, float* __restrict__ c2) {
    __shared__ float red[128];
    int row = blockIdx.x;                       // 0 .. NQ*KCB-1
    const float* p = cb + (size_t)row * DIM;
    float s = 0.f;
    for (int d = threadIdx.x; d < DIM; d += 128) { float v = p[d]; s += v * v; }
    red[threadIdx.x] = s; __syncthreads();
    for (int t = 64; t > 0; t >>= 1) {
        if ((int)threadIdx.x < t) red[threadIdx.x] += red[threadIdx.x + t];
        __syncthreads();
    }
    if (threadIdx.x == 0) c2[row] = red[0];
}

// ---------------------------------------------------------------- one VQ stage
__global__ __launch_bounds__(256) void vq_step(
        float* __restrict__ resid,
        const float* __restrict__ cb_f32,
        const unsigned short* __restrict__ cb16,
        const float* __restrict__ c2,
        float* __restrict__ lossPart,
        int* __restrict__ idx_out,
        int q) {
    struct SMemA { unsigned short aTile[NT * LDSROW]; };        // 33,280 B (bf16 residual tile)
    struct SMemB { float dMin[NT * 128]; int iMin[NT * 128]; }; // 32,768 B (overlaid after WMMA)
    union  SMemU { SMemA a; SMemB b; };
    __shared__ __align__(16) SMemU u;
    __shared__ float red[256];
    __shared__ int   selIdx[NT];

    const int tid    = threadIdx.x;
    const int token0 = blockIdx.x * NT;
    const float*          cbq   = cb_f32 + ((size_t)q << 19);   // q*1024*512
    const unsigned short* cb16q = cb16   + ((size_t)q << 19);
    const float*          c2q   = c2     + (q << 10);

    // ---- stage residual tile to LDS as bf16 (padded rows)
    for (int i = tid; i < NT * DIM; i += 256) {
        int m = i >> 9, d = i & (DIM - 1);
        float v = resid[(size_t)(token0 + m) * DIM + d];
        u.a.aTile[m * LDSROW + d] = f32_to_bf16_rne(v);
    }
    __syncthreads();

    const int wave = tid >> 5;        // 8 wave32s
    const int lane = tid & 31;
    const int half = lane >> 4;
    const int lq   = lane & 15;
    const int nBase = wave * 128;     // 128 codes per wave

    // ---- preload A fragments for both M-tiles (all 16 K-steps) into VGPRs
    const unsigned short* aRow0 = &u.a.aTile[lq * LDSROW];          // M-tile 0, row m=lq
    const unsigned short* aRow1 = &u.a.aTile[(16 + lq) * LDSROW];   // M-tile 1
    BFrag Afr[2][16];
#pragma unroll
    for (int ks = 0; ks < 16; ++ks) {
        int k0 = ks * 32;
        Afr[0][ks].h[0] = *(const ushortx8*)(aRow0 + k0 + half * 8);
        Afr[0][ks].h[1] = *(const ushortx8*)(aRow0 + k0 + 16 + half * 8);
        Afr[1][ks].h[0] = *(const ushortx8*)(aRow1 + k0 + half * 8);
        Afr[1][ks].h[1] = *(const ushortx8*)(aRow1 + k0 + 16 + half * 8);
    }

    float best[2][8];
    int   bidx[2][8];
#pragma unroll
    for (int mt = 0; mt < 2; ++mt)
#pragma unroll
        for (int v = 0; v < 8; ++v) { best[mt][v] = 3.4e38f; bidx[mt][v] = 0; }

    for (int t = 0; t < 8; ++t) {                      // 8 N-tiles of 16 codes
        int n = nBase + t * 16 + lq;
        const unsigned short* bRow = cb16q + ((size_t)n << 9);  // n*512
        v8f acc0 = {}, acc1 = {};
#pragma unroll
        for (int ks = 0; ks < 16; ++ks) {              // K = 512 in steps of 32
            int k0 = ks * 32;
            BFrag Bf;
            Bf.h[0] = *(const ushortx8*)(bRow + k0 + half * 8);        // global_load_b128
            Bf.h[1] = *(const ushortx8*)(bRow + k0 + 16 + half * 8);   // global_load_b128
            acc0 = __builtin_amdgcn_wmma_f32_16x16x32_bf16(
                       false, Afr[0][ks].v, false, Bf.v, (short)0, acc0, false, false);
            acc1 = __builtin_amdgcn_wmma_f32_16x16x32_bf16(
                       false, Afr[1][ks].v, false, Bf.v, (short)0, acc1, false, false);
        }
        float c2n = c2q[n];
#pragma unroll
        for (int v = 0; v < 8; ++v) {                  // lane: N=lq, VGPR v: M = half*8+v
            float d0 = c2n - 2.0f * acc0[v];
            if (d0 < best[0][v]) { best[0][v] = d0; bidx[0][v] = n; }
            float d1 = c2n - 2.0f * acc1[v];
            if (d1 < best[1][v]) { best[1][v] = d1; bidx[1][v] = n; }
        }
    }
    __syncthreads();   // done reading aTile -> safe to overlay dMin/iMin

    int slot = wave * 16 + lq;                          // 128 candidates per token
#pragma unroll
    for (int mt = 0; mt < 2; ++mt)
#pragma unroll
        for (int v = 0; v < 8; ++v) {
            int m = mt * 16 + half * 8 + v;
            u.b.dMin[m * 128 + slot] = best[mt][v];
            u.b.iMin[m * 128 + slot] = bidx[mt][v];
        }
    __syncthreads();

    if (tid < NT) {                                     // final argmin per token
        float bd = 3.4e38f; int bi = 0;
        for (int s = 0; s < 128; ++s) {
            float d = u.b.dMin[tid * 128 + s];
            if (d < bd) { bd = d; bi = u.b.iMin[tid * 128 + s]; }
        }
        selIdx[tid] = bi;
        idx_out[(size_t)q * NTOK + token0 + tid] = bi;
    }
    __syncthreads();

    // ---- residual update (fp32 codebook gather) + loss partial (deterministic tree reduce)
    float sq = 0.f;
    for (int i = tid; i < NT * DIM; i += 256) {
        int m = i >> 9, d = i & (DIM - 1);
        size_t gi = (size_t)(token0 + m) * DIM + d;
        float nr = resid[gi] - cbq[((size_t)selIdx[m] << 9) + d];
        resid[gi] = nr;
        sq += nr * nr;
    }
    red[tid] = sq; __syncthreads();
    for (int t = 128; t > 0; t >>= 1) {
        if (tid < t) red[tid] += red[tid + t];
        __syncthreads();
    }
    if (tid == 0) lossPart[q * NWG + blockIdx.x] = red[0];
}

// ---------------------------------------------------------------- qout = x - resid_final
__global__ void vq_finalize(const float* __restrict__ x, const float* __restrict__ resid,
                            float* __restrict__ qout) {
    size_t stride = (size_t)gridDim.x * blockDim.x;
    for (size_t i = (size_t)blockIdx.x * blockDim.x + threadIdx.x;
         i < (size_t)NTOK * DIM; i += stride)
        qout[i] = x[i] - resid[i];
}

// ---------------------------------------------------------------- losses (deterministic)
__global__ void vq_loss(const float* __restrict__ lossPart, float* __restrict__ loss_out) {
    __shared__ float red[256];
    int q = blockIdx.x;
    float s = 0.f;
    for (int i = threadIdx.x; i < NWG; i += 256) s += lossPart[q * NWG + i];
    red[threadIdx.x] = s; __syncthreads();
    for (int t = 128; t > 0; t >>= 1) {
        if ((int)threadIdx.x < t) red[threadIdx.x] += red[threadIdx.x + t];
        __syncthreads();
    }
    if (threadIdx.x == 0)
        loss_out[q] = 1.25f * red[0] / ((float)NTOK * (float)DIM);
}

// ----------------------------------------------------------------
extern "C" void kernel_launch(void* const* d_in, const int* in_sizes, int n_in,
                              void* d_out, int out_size, void* d_ws, size_t ws_size,
                              hipStream_t stream) {
    const float* x  = (const float*)d_in[0];     // [16,1500,512]
    const float* cb = (const float*)d_in[1];     // [8,1024,512]

    float* out      = (float*)d_out;
    float* qout     = out;                                          // 12,288,000 f32
    int*   idx_out  = (int*)(out + (size_t)NTOK * DIM);             // 192,000 i32
    float* loss_out = out + (size_t)NTOK * DIM + (size_t)NQ * NTOK; // 8 f32

    char* ws = (char*)d_ws;
    float*          resid    = (float*)ws;                          // 49,152,000 B
    unsigned short* cb16     = (unsigned short*)(ws + 49152000);    //  8,388,608 B
    float*          c2       = (float*)(ws + 57540608);             //     32,768 B
    float*          lossPart = (float*)(ws + 57573376);             //     24,000 B

    vq_init<<<48000, 256, 0, stream>>>(x, cb, resid, cb16);
    vq_c2<<<NQ * KCB, 128, 0, stream>>>(cb, c2);
    for (int q = 0; q < NQ; ++q)
        vq_step<<<NWG, 256, 0, stream>>>(resid, cb, cb16, c2, lossPart, idx_out, q);
    vq_finalize<<<48000, 256, 0, stream>>>(x, resid, qout);
    vq_loss<<<NQ, 256, 0, stream>>>(lossPart, loss_out);
}
